// RelTransformer_Decoder_Block_73151882985860
// MI455X (gfx1250) — matrix operand order repaired and checked by reading, not compiled, not run on hardware
//
#include <hip/hip_runtime.h>
#include <math.h>

// Problem constants (match reference)
#define Bb   2
#define Ss   4096
#define Mm   2048
#define Dd   512
#define Hh   8
#define DHd  64
#define DFFd 2048
#define EPSc 1e-6f
#define NEGc -1.0e9f

typedef __attribute__((ext_vector_type(16))) __bf16        v16bf;
typedef __attribute__((ext_vector_type(8)))  float         v8f;
typedef __attribute__((ext_vector_type(4)))  unsigned int  u32x4;
typedef __attribute__((ext_vector_type(2)))  unsigned int  u32x2;

__device__ __forceinline__ unsigned short f2bf(float f) {
    unsigned int u = __float_as_uint(f);
    u += 0x7fffu + ((u >> 16) & 1u);      // round-to-nearest-even
    return (unsigned short)(u >> 16);
}

// ---------------------------------------------------------------------------
// elementwise helpers
// ---------------------------------------------------------------------------
__global__ __launch_bounds__(256) void k_cvt_bf16(const float* __restrict__ s,
                                                  unsigned short* __restrict__ d,
                                                  long n) {
    long i = (long)blockIdx.x * blockDim.x + threadIdx.x;
    long st = (long)gridDim.x * blockDim.x;
    for (; i < n; i += st) d[i] = f2bf(s[i]);
}

__global__ __launch_bounds__(256) void k_zero_u16(unsigned short* __restrict__ d, long n) {
    long i = (long)blockIdx.x * blockDim.x + threadIdx.x;
    long st = (long)gridDim.x * blockDim.x;
    for (; i < n; i += st) d[i] = 0;
}

// sinusoid positional table -> bf16 [Ss, Dd]  (pos = S-1..0, sin|cos halves)
__global__ __launch_bounds__(256) void k_sinusoid(unsigned short* __restrict__ P) {
    int p = blockIdx.x;
    int i = threadIdx.x;                       // 0..255 = Dd/2
    float pos = (float)(Ss - 1 - p);
    float inv = powf(10000.0f, -(2.0f * (float)i) / (float)Dd);
    float ang = pos * inv;
    P[(long)p * Dd + i]        = f2bf(sinf(ang));
    P[(long)p * Dd + 256 + i]  = f2bf(cosf(ang));
}

// qu = bf16(q + u[col]); qv = bf16(q + v[col]) (qv optional)
__global__ __launch_bounds__(256) void k_bias_qbf(const float* __restrict__ q,
                                                  const float* __restrict__ u,
                                                  const float* __restrict__ v,
                                                  unsigned short* __restrict__ qu,
                                                  unsigned short* __restrict__ qv,
                                                  long n) {
    long i = (long)blockIdx.x * blockDim.x + threadIdx.x;
    long st = (long)gridDim.x * blockDim.x;
    for (; i < n; i += st) {
        int c = (int)(i & (Dd - 1));
        float base = q[i];
        qu[i] = f2bf(base + u[c]);
        if (qv) qv[i] = f2bf(base + v[c]);
    }
}

// y = bf16(a * sigmoid(b))
__global__ __launch_bounds__(256) void k_swiglu(const float* __restrict__ a,
                                                const float* __restrict__ b,
                                                unsigned short* __restrict__ y,
                                                long n) {
    long i = (long)blockIdx.x * blockDim.x + threadIdx.x;
    long st = (long)gridDim.x * blockDim.x;
    for (; i < n; i += st) {
        float bv = b[i];
        y[i] = f2bf(a[i] * (1.0f / (1.0f + __expf(-bv))));
    }
}

// out = rmsnorm(attn + resid, g) ; optional f32 + bf16 outputs. 1 wave / row.
__global__ __launch_bounds__(32) void k_rmsnorm_res(const float* __restrict__ a,
                                                    const float* __restrict__ r,
                                                    const float* __restrict__ g,
                                                    float* __restrict__ of,
                                                    unsigned short* __restrict__ ob) {
    long row = blockIdx.x;
    int lane = threadIdx.x;
    const float* ap = a + row * Dd;
    const float* rp = r + row * Dd;
    float t[16];
    float ss = 0.0f;
#pragma unroll
    for (int i = 0; i < 16; ++i) {
        int c = lane + 32 * i;
        float v = ap[c] + rp[c];            // ALPHA == 1
        t[i] = v;
        ss += v * v;
    }
#pragma unroll
    for (int o = 16; o; o >>= 1) ss += __shfl_xor(ss, o, 32);
    float iv = rsqrtf(ss * (1.0f / (float)Dd) + EPSc);
#pragma unroll
    for (int i = 0; i < 16; ++i) {
        int c = lane + 32 * i;
        float o = t[i] * g[c] * iv;
        if (of) of[row * Dd + c] = o;
        if (ob) ob[row * Dd + c] = f2bf(o);
    }
}

// ---------------------------------------------------------------------------
// bf16 WMMA GEMM:  C[M,N](f32 and/or bf16) = A[M,K](bf16) x B[K,N](bf16)
// block = 256 thr (8 waves). Tile BM=128, BN=64, BK=32. wave w -> rows 16w..16w+15.
// A staged row-major (padded rows, b128-friendly); B staged PAIR-PACKED +
// TRANSPOSED:  sBp[n][k/2] u32 = {bf16 K=2v | bf16 K=2v+1}  -> every fragment is
// two ds_load_b128 (no scalar ds_load_u16 / v_mov chains).
// M % 128 == 0, N % 64 == 0, K % 32 == 0 (guaranteed by problem sizes).
// ---------------------------------------------------------------------------
#define SA_STRIDE 40   // u16 per A row (80 B: 16B-aligned, conflict-free banks)
#define SB_STRIDE 20   // u32 per B row (80 B)

__global__ __launch_bounds__(256)
void k_gemm_bf16(const unsigned short* __restrict__ A,
                 const unsigned short* __restrict__ Bw,
                 float* __restrict__ Cf, unsigned short* __restrict__ Cb,
                 int Md, int Nd, int Kd) {
    __shared__ unsigned short sA[128 * SA_STRIDE];
    __shared__ unsigned int   sBp[64 * SB_STRIDE];
    const int bm = blockIdx.y * 128, bn = blockIdx.x * 64;
    const int tid = threadIdx.x, wave = tid >> 5, lane = tid & 31;
    const int half = lane >> 4, ln = lane & 15;
    v8f acc[4] = {};
    // A staging role: 16 contiguous u16 per thread
    const int flat = tid * 16;
    const int ra = flat >> 5, ca = flat & 31;
    // B staging role: row-pair 2rp/2rp+1, 4 columns starting at c0
    const int rp = tid >> 4, cg = tid & 15, c0 = cg * 4;
    for (int k0 = 0; k0 < Kd; k0 += 32) {
        // --- stage A (straight copy) ---
        const u32x4* sa_src = (const u32x4*)(A + (size_t)(bm + ra) * Kd + (k0 + ca));
        u32x4* sa_dst = (u32x4*)(sA + ra * SA_STRIDE + ca);
        sa_dst[0] = sa_src[0];
        sa_dst[1] = sa_src[1];
        // --- stage B (transpose + pair-pack) ---
        union { u32x2 q; unsigned short h[4]; } r0, r1;
        r0.q = *(const u32x2*)(Bw + (size_t)(k0 + 2 * rp) * Nd + (bn + c0));
        r1.q = *(const u32x2*)(Bw + (size_t)(k0 + 2 * rp + 1) * Nd + (bn + c0));
#pragma unroll
        for (int i = 0; i < 4; ++i)
            sBp[(c0 + i) * SB_STRIDE + rp] =
                (unsigned)r0.h[i] | ((unsigned)r1.h[i] << 16);
        if (k0 + 32 < Kd) {  // prefetch next K tiles toward L2
            __builtin_prefetch(A + (size_t)(bm + ra) * Kd + (k0 + 32 + ca), 0, 1);
            __builtin_prefetch(Bw + (size_t)(k0 + 32 + 2 * rp) * Nd + (bn + c0), 0, 1);
        }
        __syncthreads();
        // A fragment 16x32 (lanes<16 K {0..7,16..23}; lanes>=16 +8) : 2x b128
        union { v16bf v; u32x4 q[2]; } fa;
        const unsigned short* arow = sA + (wave * 16 + ln) * SA_STRIDE + half * 8;
        fa.q[0] = *(const u32x4*)(arow);
        fa.q[1] = *(const u32x4*)(arow + 16);
#pragma unroll
        for (int sub = 0; sub < 4; ++sub) {
            // B fragment 32x16 (N = ln, K pairs contiguous) : 2x b128
            union { v16bf v; u32x4 q[2]; } fb;
            const unsigned int* brow = sBp + (sub * 16 + ln) * SB_STRIDE + half * 4;
            fb.q[0] = *(const u32x4*)(brow);
            fb.q[1] = *(const u32x4*)(brow + 8);
            acc[sub] = __builtin_amdgcn_wmma_f32_16x16x32_bf16(
                false, fa.v, false, fb.v, (short)0, acc[sub], false, false);
        }
        __syncthreads();
    }
#pragma unroll
    for (int sub = 0; sub < 4; ++sub) {
#pragma unroll
        for (int g = 0; g < 8; ++g) {
            int row = bm + wave * 16 + g + half * 8;
            int col = bn + sub * 16 + ln;
            float v = acc[sub][g];
            size_t o = (size_t)row * Nd + col;
            if (Cf) Cf[o] = v;
            if (Cb) Cb[o] = f2bf(v);
        }
    }
}

// ---------------------------------------------------------------------------
// Flash attention with optional Transformer-XL bd term.
// One wave32 per (b, h, 16-query tile). K-step = 32 keys.
//   scores = (qu.K^T + rel_shift(qv.R^T)) * scale, causal+len mask, online softmax,
//   O += P.V via WMMA.  bd gather uses ds_bpermute on WMMA C-layout diagonals.
// V tile staged pair-packed+transposed (u32 = {V[2j][d] | V[2j+1][d]}) so each
// V fragment is two ds_load_b128.
// ---------------------------------------------------------------------------
__global__ __launch_bounds__(32)
void k_flash(const unsigned short* __restrict__ qu,
             const unsigned short* __restrict__ qv,
             const unsigned short* __restrict__ kb,
             const unsigned short* __restrict__ vb,
             const unsigned short* __restrict__ Rb,   // [Ss+32, Dd] zero-padded, or null
             const int* __restrict__ lens,
             float* __restrict__ out,
             int Sq, int Lk, int causal, float scale) {
    __shared__ unsigned short pl[16 * 36];        // P tile 16x32 bf16
    __shared__ unsigned int   vlp[64 * SB_STRIDE]; // V tile pair-packed [d][dk/2]
    const int QT = Sq >> 4;
    int bx = blockIdx.x;
    int qt = bx % QT;
    int h  = (bx / QT) % Hh;
    int b  = bx / (QT * Hh);
    int q0 = qt << 4;
    int lane = threadIdx.x, half = lane >> 4, ln = lane & 15;
    int len = lens[b]; if (len > Lk) len = Lk;
    int hd = h * DHd;

    // load q fragments (A layout, K=64 -> two 32-chunks)
    v16bf aq[2], av[2];
    {
        const unsigned short* qrow = qu + ((size_t)b * Sq + q0 + ln) * Dd + hd + half * 8;
#pragma unroll
        for (int c = 0; c < 2; ++c) {
            union { v16bf v; u32x4 q[2]; } t;
            t.q[0] = *(const u32x4*)(qrow + c * 32);
            t.q[1] = *(const u32x4*)(qrow + c * 32 + 16);
            aq[c] = t.v;
        }
        if (qv) {
            const unsigned short* q2 = qv + ((size_t)b * Sq + q0 + ln) * Dd + hd + half * 8;
#pragma unroll
            for (int c = 0; c < 2; ++c) {
                union { v16bf v; u32x4 q[2]; } t;
                t.q[0] = *(const u32x4*)(q2 + c * 32);
                t.q[1] = *(const u32x4*)(q2 + c * 32 + 16);
                av[c] = t.v;
            }
        } else {
            av[0] = aq[0]; av[1] = aq[1];
        }
    }

    float mrow[8], lrow[8];
    v8f accO[4] = {};
#pragma unroll
    for (int g = 0; g < 8; ++g) { mrow[g] = -3.0e38f; lrow[g] = 0.0f; }

    int kend = causal ? (q0 + 16) : ((len + 31) & ~31);
    for (int k0 = 0; k0 < kend; k0 += 32) {
        // --- stage V tile: lane (half,ln) owns dk rows {2ln,2ln+1}, d cols half*32.. ---
        {
            int r0i = k0 + 2 * ln;      if (r0i > Lk - 1) r0i = Lk - 1;
            int r1i = k0 + 2 * ln + 1;  if (r1i > Lk - 1) r1i = Lk - 1;
            int dbase = half * 32;
            const u32x4* p0 = (const u32x4*)(vb + ((size_t)b * Lk + r0i) * Dd + hd + dbase);
            const u32x4* p1 = (const u32x4*)(vb + ((size_t)b * Lk + r1i) * Dd + hd + dbase);
#pragma unroll
            for (int c = 0; c < 4; ++c) {
                union { u32x4 q; unsigned short h[8]; } a0, a1;
                a0.q = p0[c];
                a1.q = p1[c];
#pragma unroll
                for (int i = 0; i < 8; ++i)
                    vlp[(dbase + c * 8 + i) * SB_STRIDE + ln] =
                        (unsigned)a0.h[i] | ((unsigned)a1.h[i] << 16);
            }
        }
        float sc[2][8];
#pragma unroll
        for (int st = 0; st < 2; ++st) {
            int kt = k0 + st * 16;
            // ac = qu . K^T  (B frag straight from global: K {0..7,16..23}, +8 for half 1)
            v8f aS = {};
#pragma unroll
            for (int c = 0; c < 2; ++c) {
                int kr = kt + ln; if (kr > Lk - 1) kr = Lk - 1;
                const unsigned short* krow =
                    kb + ((size_t)b * Lk + kr) * Dd + hd + c * 32 + half * 8;
                union { v16bf v; u32x4 q[2]; } t;
                t.q[0] = *(const u32x4*)(krow);
                t.q[1] = *(const u32x4*)(krow + 16);
                aS = __builtin_amdgcn_wmma_f32_16x16x32_bf16(
                    false, aq[c], false, t.v, (short)0, aS, false, false);
            }
            float bdv[8] = {0, 0, 0, 0, 0, 0, 0, 0};
            if (Rb) {
                // banded G[dq,dj] = qv . r[jb-15+dj], dj in [0,31]; bd = diag gather
                int jb = kt - q0 + Sq - 1;
                v8f gac0 = {}, gac1 = {};
#pragma unroll
                for (int gh = 0; gh < 2; ++gh) {
                    v8f ga = {};
#pragma unroll
                    for (int c = 0; c < 2; ++c) {
                        int jr = jb - 15 + gh * 16 + ln;   // in [0, Ss+31], pad rows are 0
                        const unsigned short* rrow =
                            Rb + (size_t)jr * Dd + hd + c * 32 + half * 8;
                        union { v16bf v; u32x4 q[2]; } t;
                        t.q[0] = *(const u32x4*)(rrow);
                        t.q[1] = *(const u32x4*)(rrow + 16);
                        ga = __builtin_amdgcn_wmma_f32_16x16x32_bf16(
                            false, av[c], false, t.v, (short)0, ga, false, false);
                    }
                    if (gh == 0) gac0 = ga; else gac1 = ga;
                }
#pragma unroll
                for (int g = 0; g < 8; ++g) {
                    int dq = g + half * 8;
                    int dj = ln - dq + 15;                 // 0..30
                    int scol = dj & 15, ssel = dj >> 4;
                    int idx = (scol + half * 16) << 2;     // bpermute index in bytes
                    float g0 = __int_as_float(
                        __builtin_amdgcn_ds_bpermute(idx, __float_as_int(gac0[g])));
                    float g1 = __int_as_float(
                        __builtin_amdgcn_ds_bpermute(idx, __float_as_int(gac1[g])));
                    bdv[g] = ssel ? g1 : g0;
                }
            }
#pragma unroll
            for (int g = 0; g < 8; ++g) {
                int dq = g + half * 8;
                int qi = q0 + dq, ki = kt + ln;
                float s = (aS[g] + bdv[g]) * scale;
                bool ok = (ki < len) && (!causal || ki <= qi);
                sc[st][g] = ok ? s : NEGc;
            }
        }
        // online softmax over the 32 new keys (row = 16 lanes of one half)
#pragma unroll
        for (int g = 0; g < 8; ++g) {
            float mt = fmaxf(sc[0][g], sc[1][g]);
#pragma unroll
            for (int o = 8; o; o >>= 1) mt = fmaxf(mt, __shfl_xor(mt, o, 16));
            float mn = fmaxf(mrow[g], mt);
            float corr = __expf(mrow[g] - mn);
            float p0 = __expf(sc[0][g] - mn);
            float p1 = __expf(sc[1][g] - mn);
            float rs = p0 + p1;
#pragma unroll
            for (int o = 8; o; o >>= 1) rs += __shfl_xor(rs, o, 16);
            lrow[g] = lrow[g] * corr + rs;
            mrow[g] = mn;
#pragma unroll
            for (int t2 = 0; t2 < 4; ++t2) accO[t2][g] *= corr;
            int dq = g + half * 8;
            pl[dq * 36 + ln]      = f2bf(p0);
            pl[dq * 36 + 16 + ln] = f2bf(p1);
        }
        asm volatile("s_wait_dscnt 0" ::: "memory");   // wave-private LDS sync
        // P as A fragment (16x32) : 2x b128
        union { v16bf v; u32x4 q[2]; } pf;
        {
            const unsigned short* prow = pl + ln * 36 + half * 8;
            pf.q[0] = *(const u32x4*)(prow);
            pf.q[1] = *(const u32x4*)(prow + 16);
        }
        // O += P . V  (V fragments: 2x b128 from pair-packed LDS)
#pragma unroll
        for (int sub = 0; sub < 4; ++sub) {
            union { v16bf v; u32x4 q[2]; } fb;
            const unsigned int* brow = vlp + (sub * 16 + ln) * SB_STRIDE + half * 4;
            fb.q[0] = *(const u32x4*)(brow);
            fb.q[1] = *(const u32x4*)(brow + 8);
            accO[sub] = __builtin_amdgcn_wmma_f32_16x16x32_bf16(
                false, pf.v, false, fb.v, (short)0, accO[sub], false, false);
        }
    }
    // epilogue: divide by l, scatter to [B,S,D] f32 head slice
#pragma unroll
    for (int sub = 0; sub < 4; ++sub) {
#pragma unroll
        for (int g = 0; g < 8; ++g) {
            int row = q0 + g + half * 8;
            int col = hd + sub * 16 + ln;
            out[((size_t)b * Sq + row) * Dd + col] = accO[sub][g] / lrow[g];
        }
    }
}

// ---------------------------------------------------------------------------
// host orchestration
// ---------------------------------------------------------------------------
extern "C" void kernel_launch(void* const* d_in, const int* in_sizes, int n_in,
                              void* d_out, int out_size, void* d_ws, size_t ws_size,
                              hipStream_t stream) {
    (void)in_sizes; (void)n_in; (void)out_size; (void)ws_size;
    const float* x       = (const float*)d_in[0];
    const float* ctx     = (const float*)d_in[1];
    const int*   tgt_len = (const int*)d_in[2];
    const int*   ctx_len = (const int*)d_in[3];
    const float* u0 = (const float*)d_in[4];
    const float* v0 = (const float*)d_in[5];
    const float* u1 = (const float*)d_in[6];
    /* v1 = d_in[7] is unused by the reference cross-attention */
    const float* Wq0 = (const float*)d_in[8];
    const float* Wk0 = (const float*)d_in[9];
    const float* Wv0 = (const float*)d_in[10];
    const float* Wo0 = (const float*)d_in[11];
    const float* Wr  = (const float*)d_in[12];
    const float* Wq1 = (const float*)d_in[13];
    const float* Wk1 = (const float*)d_in[14];
    const float* Wv1 = (const float*)d_in[15];
    const float* Wo1 = (const float*)d_in[16];
    const float* g1  = (const float*)d_in[17];
    const float* g2  = (const float*)d_in[18];
    const float* g3  = (const float*)d_in[19];
    const float* Wa  = (const float*)d_in[20];
    const float* Wb  = (const float*)d_in[21];
    const float* Wc  = (const float*)d_in[22];
    float* outp = (float*)d_out;

    char* base = (char*)d_ws;
    size_t off = 0;
    auto alloc = [&](size_t bytes) -> void* {
        size_t a = (off + 255) & ~(size_t)255;
        off = a + bytes;
        return (void*)(base + a);
    };

    const size_t BSD = (size_t)Bb * Ss * Dd;
    const size_t BMD = (size_t)Bb * Mm * Dd;
    const size_t DDs = (size_t)Dd * Dd;
    const size_t DFs = (size_t)Dd * DFFd;
    const size_t SD  = (size_t)Ss * Dd;
    const size_t RE  = (size_t)(Ss + 32) * Dd;
    const size_t FF  = (size_t)Bb * Ss * DFFd;

    unsigned short* xb   = (unsigned short*)alloc(BSD * 2);
    unsigned short* cbuf = (unsigned short*)alloc(BMD * 2);
    unsigned short* wq0b = (unsigned short*)alloc(DDs * 2);
    unsigned short* wk0b = (unsigned short*)alloc(DDs * 2);
    unsigned short* wv0b = (unsigned short*)alloc(DDs * 2);
    unsigned short* wo0b = (unsigned short*)alloc(DDs * 2);
    unsigned short* wrb  = (unsigned short*)alloc(DDs * 2);
    unsigned short* wq1b = (unsigned short*)alloc(DDs * 2);
    unsigned short* wk1b = (unsigned short*)alloc(DDs * 2);
    unsigned short* wv1b = (unsigned short*)alloc(DDs * 2);
    unsigned short* wo1b = (unsigned short*)alloc(DDs * 2);
    unsigned short* wab  = (unsigned short*)alloc(DFs * 2);
    unsigned short* wbb  = (unsigned short*)alloc(DFs * 2);
    unsigned short* wcb  = (unsigned short*)alloc(DFs * 2);
    unsigned short* psin = (unsigned short*)alloc(SD * 2);
    unsigned short* Rb   = (unsigned short*)alloc(RE * 2);
    unsigned short* qub  = (unsigned short*)alloc(BSD * 2);
    unsigned short* qvb  = (unsigned short*)alloc(BSD * 2);
    unsigned short* kbs  = (unsigned short*)alloc(BSD * 2);
    unsigned short* vbs  = (unsigned short*)alloc(BSD * 2);
    unsigned short* sab  = (unsigned short*)alloc(BSD * 2);
    unsigned short* x1b  = (unsigned short*)alloc(BSD * 2);
    unsigned short* q1b  = (unsigned short*)alloc(BSD * 2);
    unsigned short* kcb  = (unsigned short*)alloc(BMD * 2);
    unsigned short* vcb  = (unsigned short*)alloc(BMD * 2);
    unsigned short* cab  = (unsigned short*)alloc(BSD * 2);
    unsigned short* x2b  = (unsigned short*)alloc(BSD * 2);
    unsigned short* yb   = (unsigned short*)alloc(FF * 2);
    float* t1  = (float*)alloc(FF * 4);
    float* t2  = (float*)alloc(FF * 4);
    float* x1f = (float*)alloc(BSD * 4);
    float* x2f = (float*)alloc(BSD * 4);

    auto cvt = [&](const float* s, unsigned short* d, size_t n) {
        k_cvt_bf16<<<2048, 256, 0, stream>>>(s, d, (long)n);
    };
    auto gemm = [&](const unsigned short* A, const unsigned short* Bm,
                    float* Cf, unsigned short* Cb2, int Md, int Nd, int Kd) {
        dim3 grid(Nd / 64, Md / 128);
        k_gemm_bf16<<<grid, 256, 0, stream>>>(A, Bm, Cf, Cb2, Md, Nd, Kd);
    };

    // --- precision conversion ---
    cvt(x, xb, BSD);
    cvt(ctx, cbuf, BMD);
    cvt(Wq0, wq0b, DDs); cvt(Wk0, wk0b, DDs); cvt(Wv0, wv0b, DDs); cvt(Wo0, wo0b, DDs);
    cvt(Wr, wrb, DDs);
    cvt(Wq1, wq1b, DDs); cvt(Wk1, wk1b, DDs); cvt(Wv1, wv1b, DDs); cvt(Wo1, wo1b, DDs);
    cvt(Wa, wab, DFs); cvt(Wb, wbb, DFs); cvt(Wc, wcb, DFs);

    // --- relative position table R = sinusoid @ Wr (zero-padded tail rows) ---
    k_sinusoid<<<Ss, 256, 0, stream>>>(psin);
    gemm(psin, wrb, nullptr, Rb, Ss, Dd, Dd);
    k_zero_u16<<<64, 256, 0, stream>>>(Rb + (size_t)Ss * Dd, (long)(32 * Dd));

    // --- self attention (XL relative) ---
    gemm(xb, wq0b, t1, nullptr, Bb * Ss, Dd, Dd);
    k_bias_qbf<<<2048, 256, 0, stream>>>(t1, u0, v0, qub, qvb, (long)BSD);
    gemm(xb, wk0b, nullptr, kbs, Bb * Ss, Dd, Dd);
    gemm(xb, wv0b, nullptr, vbs, Bb * Ss, Dd, Dd);
    k_flash<<<Bb * Hh * (Ss / 16), 32, 0, stream>>>(
        qub, qvb, kbs, vbs, Rb, tgt_len, t2, Ss, Ss, 1, 0.125f);
    cvt(t2, sab, BSD);
    gemm(sab, wo0b, t1, nullptr, Bb * Ss, Dd, Dd);
    k_rmsnorm_res<<<Bb * Ss, 32, 0, stream>>>(t1, x, g1, x1f, x1b);

    // --- cross attention (u-bias only) ---
    gemm(x1b, wq1b, t1, nullptr, Bb * Ss, Dd, Dd);
    k_bias_qbf<<<2048, 256, 0, stream>>>(t1, u1, nullptr, q1b, nullptr, (long)BSD);
    gemm(cbuf, wk1b, nullptr, kcb, Bb * Mm, Dd, Dd);
    gemm(cbuf, wv1b, nullptr, vcb, Bb * Mm, Dd, Dd);
    k_flash<<<Bb * Hh * (Ss / 16), 32, 0, stream>>>(
        q1b, nullptr, kcb, vcb, nullptr, ctx_len, t2, Ss, Mm, 0, 0.125f);
    cvt(t2, cab, BSD);
    gemm(cab, wo1b, t1, nullptr, Bb * Ss, Dd, Dd);
    k_rmsnorm_res<<<Bb * Ss, 32, 0, stream>>>(t1, x1f, g2, x2f, x2b);

    // --- GLU FFN ---
    gemm(x2b, wab, t1, nullptr, Bb * Ss, DFFd, Dd);
    gemm(x2b, wbb, t2, nullptr, Bb * Ss, DFFd, Dd);
    k_swiglu<<<4096, 256, 0, stream>>>(t1, t2, yb, (long)FF);
    gemm(yb, wcb, t1, nullptr, Bb * Ss, Dd, DFFd);
    k_rmsnorm_res<<<Bb * Ss, 32, 0, stream>>>(t1, x2f, g3, outp, nullptr);
}
